// ClusteringLayer_43207370998133
// MI455X (gfx1250) — matrix operand order repaired and checked by reading, not compiled, not run on hardware
//
#include <hip/hip_runtime.h>
#include <hip/hip_bf16.h>

typedef __attribute__((ext_vector_type(16))) _Float16 v16h;
typedef __attribute__((ext_vector_type(8)))  float    v8f;
typedef __attribute__((ext_vector_type(4)))  float    v4f;

#define DDIM 256          // feature dim
#define KCL  256          // clusters
#define BROWS 64          // rows per block

__device__ __forceinline__ unsigned int pack_f16x2(float a, float b) {
    _Float16 ha = (_Float16)a, hb = (_Float16)b;
    unsigned short ua = __builtin_bit_cast(unsigned short, ha);
    unsigned short ub = __builtin_bit_cast(unsigned short, hb);
    return (unsigned int)ua | ((unsigned int)ub << 16);
}

// ---------------------------------------------------------------------------
// c2[k] = sum_d clusters[k][d]^2   (fp32, exact)
// ---------------------------------------------------------------------------
__global__ __launch_bounds__(256) void k_c2(const float* __restrict__ clusters,
                                            float* __restrict__ c2) {
    int k = threadIdx.x;
    const float* row = clusters + (size_t)k * DDIM;
    float s = 0.f;
    #pragma unroll 8
    for (int d = 0; d < DDIM; ++d) { float v = row[d]; s += v * v; }
    c2[k] = s;
}

// ---------------------------------------------------------------------------
// Repack clusters^T into f16 WMMA B-fragments.
// Fragment (kk, nt) covers K-slice d0=kk*32 (32 deep) x cols nt*16..+15.
// B layout (32x16, f16): lanes 0-15: V0-3 -> K0..7, V4-7 -> K16..23;
//                        lanes 16-31: V0-3 -> K8..15, V4-7 -> K24..31.
// Lane l's 16 halves stored contiguously: bfrag[(kk*16+nt)*512 + l*16 + j].
// grid = 128 blocks (kk*16+nt), 256 threads: t -> l = t/8, vg = t%8.
// ---------------------------------------------------------------------------
__global__ __launch_bounds__(256) void k_bfrag(const float* __restrict__ clusters,
                                               unsigned int* __restrict__ bfrag32) {
    int b  = blockIdx.x;           // kk*16 + nt
    int kk = b >> 4, nt = b & 15;
    int t  = threadIdx.x;
    int l  = t >> 3, vg = t & 7;
    int half = l >> 4;
    int n = nt * 16 + (l & 15);
    int d = kk * 32 + (vg >> 2) * 16 + half * 8 + (vg & 3) * 2;
    float a = clusters[(size_t)n * DDIM + d];
    float c = clusters[(size_t)n * DDIM + d + 1];
    bfrag32[(size_t)b * 256 + t] = pack_f16x2(a, c);
}

// ---------------------------------------------------------------------------
// Main fused kernel: GEMM (f16 WMMA, f32 acc) + dist2 + q + row-normalize.
// ---------------------------------------------------------------------------
__global__ __launch_bounds__(256) void k_main(const float* __restrict__ x,
                                              const float* __restrict__ c2,
                                              const _Float16* __restrict__ bfrag,
                                              float* __restrict__ out) {
    // A-fragments for 8 k-steps x 4 m-tiles, 512 halves each = 32KB
    __shared__ __align__(32) _Float16 xfrag[8 * 4 * 512];
    __shared__ float x2s[BROWS];
    __shared__ float rowsum[BROWS];

    const int t    = threadIdx.x;
    const int w    = t >> 5;        // wave 0..7
    const int lane = t & 31;
    const size_t rowbase = (size_t)blockIdx.x * BROWS;

    if (t < BROWS) { x2s[t] = 0.f; rowsum[t] = 0.f; }
    __syncthreads();

    // ---- stage x tile: fp32 -> f16 A-fragments in LDS, accumulate x^2 ----
    // A layout (16x32 f16): lanes 0-15: V0-3 -> K0..7, V4-7 -> K16..23;
    //                       lanes16-31: V0-3 -> K8..15, V4-7 -> K24..31.
    {
        const v4f* xv = (const v4f*)(x + rowbase * DDIM);
        unsigned int* xf32 = (unsigned int*)xfrag;
        #pragma unroll
        for (int i = 0; i < 16; ++i) {
            int f   = i * 256 + t;       // float4 index in tile
            int row = f >> 6;            // local row 0..63
            int d   = (f & 63) << 2;     // feature 0..252 step 4
            v4f v = __builtin_nontemporal_load(&xv[f]);
            atomicAdd(&x2s[row], v.x * v.x + v.y * v.y + v.z * v.z + v.w * v.w);
            int kk = d >> 5, kl = d & 31;
            int group = kl >> 4, rem = kl & 15;
            int half = rem >> 3, idx2 = rem & 7;
            int vg = group * 4 + (idx2 >> 1);
            int mt = row >> 4, rm = row & 15;
            int l  = half * 16 + rm;
            int idx = (kk * 4 + mt) * 256 + l * 8 + vg;   // 32-bit units
            xf32[idx]     = pack_f16x2(v.x, v.y);
            xf32[idx + 1] = pack_f16x2(v.z, v.w);
        }
    }
    __syncthreads();

    // ---- GEMM: each wave: 4 m-tiles x 2 n-tiles, K loop of 8 steps ----
    v8f acc[4][2];
    {
        v8f z = {0.f, 0.f, 0.f, 0.f, 0.f, 0.f, 0.f, 0.f};
        #pragma unroll
        for (int mt = 0; mt < 4; ++mt) { acc[mt][0] = z; acc[mt][1] = z; }
    }
    const _Float16* bp0 = bfrag + (size_t)(2 * w)     * 512 + lane * 16;
    const _Float16* bp1 = bfrag + (size_t)(2 * w + 1) * 512 + lane * 16;
    #pragma unroll
    for (int kk = 0; kk < 8; ++kk) {
        v16h b0 = *(const v16h*)(bp0 + (size_t)kk * 16 * 512);
        v16h b1 = *(const v16h*)(bp1 + (size_t)kk * 16 * 512);
        #pragma unroll
        for (int mt = 0; mt < 4; ++mt) {
            v16h a = *(const v16h*)(xfrag + (kk * 4 + mt) * 512 + lane * 16);
            acc[mt][0] = __builtin_amdgcn_wmma_f32_16x16x32_f16(
                false, a, false, b0, (short)0, acc[mt][0], false, false);
            acc[mt][1] = __builtin_amdgcn_wmma_f32_16x16x32_f16(
                false, a, false, b1, (short)0, acc[mt][1], false, false);
        }
    }

    // ---- epilogue: dist2 -> q -> row sums (shfl + LDS atomics) ----
    // C layout: lane half h, VGPR v -> row = 16*mt + 8*h + v, col = ntile*16 + (lane&15)
    const int half = lane >> 4;
    const int col0 = w * 32 + (lane & 15);
    const float c2a = c2[col0];
    const float c2b = c2[col0 + 16];
    #pragma unroll
    for (int mt = 0; mt < 4; ++mt) {
        #pragma unroll
        for (int v = 0; v < 8; ++v) {
            int rloc = mt * 16 + half * 8 + v;
            float x2r = x2s[rloc];
            float q0 = 1.f / (1.f + x2r + c2a - 2.f * acc[mt][0][v]);
            float q1 = 1.f / (1.f + x2r + c2b - 2.f * acc[mt][1][v]);
            acc[mt][0][v] = q0;
            acc[mt][1][v] = q1;
            float p = q0 + q1;                 // sum over this wave's 2 n-tiles
            p += __shfl_xor(p, 1);             // reduce over 16 cols (stays in half)
            p += __shfl_xor(p, 2);
            p += __shfl_xor(p, 4);
            p += __shfl_xor(p, 8);
            if ((lane & 15) == 0) atomicAdd(&rowsum[rloc], p);
        }
    }
    __syncthreads();

    // ---- normalize + store ----
    #pragma unroll
    for (int mt = 0; mt < 4; ++mt) {
        #pragma unroll
        for (int v = 0; v < 8; ++v) {
            int rloc = mt * 16 + half * 8 + v;
            float inv = 1.f / rowsum[rloc];
            float* orow = out + (rowbase + (size_t)rloc) * KCL;
            __builtin_nontemporal_store(acc[mt][0][v] * inv, &orow[col0]);
            __builtin_nontemporal_store(acc[mt][1][v] * inv, &orow[col0 + 16]);
        }
    }
}

// ---------------------------------------------------------------------------
extern "C" void kernel_launch(void* const* d_in, const int* in_sizes, int n_in,
                              void* d_out, int out_size, void* d_ws, size_t ws_size,
                              hipStream_t stream) {
    const float* x        = (const float*)d_in[0];   // (131072, 256) fp32
    const float* clusters = (const float*)d_in[1];   // (256, 256) fp32
    float* out = (float*)d_out;                      // (131072, 256) fp32

    float*    c2    = (float*)d_ws;                          // 1 KB
    _Float16* bfrag = (_Float16*)((char*)d_ws + 1024);       // 128 KB

    const int B = in_sizes[0] / DDIM;                // 131072

    k_c2   <<<1,   256, 0, stream>>>(clusters, c2);
    k_bfrag<<<128, 256, 0, stream>>>(clusters, (unsigned int*)bfrag);
    k_main <<<B / BROWS, 256, 0, stream>>>(x, c2, bfrag, out);
}
